// RetrievalAugmentedModule_57329223467516
// MI455X (gfx1250) — matrix operand order repaired and checked by reading, not compiled, not run on hardware
//
#include <hip/hip_runtime.h>
#include <hip/hip_bf16.h>

// ---------------------------------------------------------------------------
// MI455X (gfx1250) fused retrieval-augmented module.
//  - All GEMMs via v_wmma_f32_16x16x32_bf16 (fp32 -> bf16 inputs, f32 accum)
//  - sim (65536x4096) never materialized: fused top-8 + softmax + gather
//  - keys streamed via GLOBAL_LOAD_ASYNC_TO_LDS_B128, double buffered
// ---------------------------------------------------------------------------

typedef __bf16 bf16_t;
typedef __attribute__((ext_vector_type(8)))  float  v8f;
typedef __attribute__((ext_vector_type(8)))  bf16_t v8bf;
typedef __attribute__((ext_vector_type(16))) bf16_t v16bf;

typedef int v4i __attribute__((vector_size(16)));
typedef __attribute__((address_space(1))) v4i* gv4i_p;   // global int4*
typedef __attribute__((address_space(3))) v4i* lv4i_p;   // LDS int4*

#if __has_builtin(__builtin_amdgcn_global_load_async_to_lds_b128)
#define HAVE_ASYNC_LDS 1
#else
#define HAVE_ASYNC_LDS 0
#endif

// 16-byte global -> LDS copy. Async (ASYNCcnt-tracked, no VGPR transit) when
// the CDNA5 builtin is available; plain load/store otherwise.
static __device__ __forceinline__ void cp16_g2lds(void* dst_lds, const void* src_global) {
#if HAVE_ASYNC_LDS
  __builtin_amdgcn_global_load_async_to_lds_b128(
      (gv4i_p)(v4i*)const_cast<void*>(src_global),
      (lv4i_p)(v4i*)dst_lds,
      0, 0);
#else
  *(uint4*)dst_lds = *(const uint4*)src_global;
#endif
}

static __device__ __forceinline__ void async_wait0() {
#if HAVE_ASYNC_LDS
#if __has_builtin(__builtin_amdgcn_s_wait_asynccnt)
  __builtin_amdgcn_s_wait_asynccnt(0);
#else
  asm volatile("s_wait_asynccnt 0x0" ::: "memory");
#endif
#endif
}

static __device__ __forceinline__ v8f wmma_bf16(v16bf a, v16bf b, v8f c) {
  // D = A(16x32 bf16) x B(32x16 bf16) + C(16x16 f32)
  return __builtin_amdgcn_wmma_f32_16x16x32_bf16(
      false, a, false, b, (short)0, c, false, false);
}

// B-tile per-lane fragment: bf16 chunks [p..p+7] and [p+16..p+23] (same K order as A).
static __device__ __forceinline__ v16bf load_bf16x16(const bf16_t* p) {
  v8bf lo = *(const v8bf*)(p);
  v8bf hh = *(const v8bf*)(p + 16);
  v16bf r;
#pragma unroll
  for (int i = 0; i < 8; ++i) { r[i] = lo[i]; r[i + 8] = hh[i]; }
  return r;
}

// A-tile per-lane fragment: bf16 chunks [p..p+7] and [p+16..p+23]
// (ISA 16x32 16-bit A layout: VGPR0-3 = first 8 Ks, VGPR4-7 = +16).
static __device__ __forceinline__ v16bf load_a_lds(const bf16_t* p) {
  v8bf lo = *(const v8bf*)(p);
  v8bf hh = *(const v8bf*)(p + 16);
  v16bf r;
#pragma unroll
  for (int i = 0; i < 8; ++i) { r[i] = lo[i]; r[i + 8] = hh[i]; }
  return r;
}

// A fragment built directly from fp32 global memory (convert to bf16).
static __device__ __forceinline__ v16bf load_a_f32(const float* p) {
  float4 f0 = *(const float4*)(p);
  float4 f1 = *(const float4*)(p + 4);
  float4 f2 = *(const float4*)(p + 16);
  float4 f3 = *(const float4*)(p + 20);
  v16bf r;
  r[0]=(bf16_t)f0.x; r[1]=(bf16_t)f0.y; r[2]=(bf16_t)f0.z; r[3]=(bf16_t)f0.w;
  r[4]=(bf16_t)f1.x; r[5]=(bf16_t)f1.y; r[6]=(bf16_t)f1.z; r[7]=(bf16_t)f1.w;
  r[8]=(bf16_t)f2.x; r[9]=(bf16_t)f2.y; r[10]=(bf16_t)f2.z; r[11]=(bf16_t)f2.w;
  r[12]=(bf16_t)f3.x; r[13]=(bf16_t)f3.y; r[14]=(bf16_t)f3.z; r[15]=(bf16_t)f3.w;
  return r;
}

// ------------------------- prep kernels -------------------------------------

__global__ void k_cvt_bf16(const float* __restrict__ in, bf16_t* __restrict__ out, int n) {
  int i = blockIdx.x * blockDim.x + threadIdx.x;
  if (i < n) out[i] = (bf16_t)in[i];
}

// out[c*rows + r] = in[r*cols + c]   (fp32 -> bf16, transpose to [out][in])
__global__ void k_transpose_bf16(const float* __restrict__ in, bf16_t* __restrict__ out,
                                 int rows, int cols) {
  int i = blockIdx.x * blockDim.x + threadIdx.x;
  if (i < rows * cols) {
    int r = i / cols, c = i - r * cols;
    out[c * rows + r] = (bf16_t)in[i];
  }
}

// ------------------------- fused retrieve kernel ----------------------------
// grid: 512 blocks x 256 threads (8 waves); 128 query rows per block.

__global__ __launch_bounds__(256)
void k_retrieve(const float* __restrict__ query,
                const bf16_t* __restrict__ keys_bf,   // [4096][256]
                const float* __restrict__ values,     // [4096][256] f32
                const bf16_t* __restrict__ WqT,       // [256][256] = Wq^T
                const float* __restrict__ bq,
                bf16_t* __restrict__ retrieved_bf)    // [65536][256]
{
  constexpr int LDK = 264;   // padded row stride (bf16 elems) -> bank spread
  __shared__ bf16_t qbf[128 * LDK];   // 67584 B : projected q, bf16
  __shared__ char   scratch[92672];   // overlaid phase regions

  const int tid  = threadIdx.x;
  const int wave = tid >> 5;
  const int lane = tid & 31;
  const int lhi  = lane >> 4;     // lane-half (WMMA A/B/C layout split)
  const int lcol = lane & 15;
  const int rowBlock = blockIdx.x * 128;
  const int mloc = wave * 16;

  // ---------------- phase 1: q = query @ Wq + bq ----------------
  bf16_t* wqlds = (bf16_t*)scratch;  // 64 rows of WqT per chunk (33792 B)

  v16bf a1[8];
  {
    const float* rp = query + (size_t)(rowBlock + mloc + lcol) * 256;
#pragma unroll
    for (int kc = 0; kc < 8; ++kc)
      a1[kc] = load_a_f32(rp + kc * 32 + (lhi ? 8 : 0));
  }

  for (int nch = 0; nch < 4; ++nch) {
    for (int i = tid; i < 64 * 32; i += 256) {          // 64 rows x 32 x 16B
      int r = i >> 5, c = i & 31;
      cp16_g2lds(wqlds + r * LDK + c * 8,
                 WqT + (size_t)(nch * 64 + r) * 256 + c * 8);
    }
    async_wait0();
    __syncthreads();
#pragma unroll
    for (int t = 0; t < 4; ++t) {
      v8f acc = {0.f,0.f,0.f,0.f,0.f,0.f,0.f,0.f};
#pragma unroll
      for (int kc = 0; kc < 8; ++kc) {
        v16bf b = load_bf16x16(wqlds + (t * 16 + lcol) * LDK + kc * 32 + (lhi ? 8 : 0));
        acc = wmma_bf16(a1[kc], b, acc);
      }
      int ncol = nch * 64 + t * 16 + lcol;
      float bias = bq[ncol];
#pragma unroll
      for (int j = 0; j < 8; ++j)
        qbf[(mloc + j + 8 * lhi) * LDK + ncol] = (bf16_t)(acc[j] + bias);
    }
    __syncthreads();
  }

  // ---------------- phase 2: sim + fused top-8 ----------------
  v16bf a2[8];
  {
    const bf16_t* rp = qbf + (mloc + lcol) * LDK;
#pragma unroll
    for (int kc = 0; kc < 8; ++kc)
      a2[kc] = load_a_lds(rp + kc * 32 + (lhi ? 8 : 0));
  }

  // double-buffered 64-row keys chunks (2 x 33792 B in same region)
  bf16_t*   kbuf0   = (bf16_t*)scratch;
  bf16_t*   kbuf1   = (bf16_t*)scratch + 64 * LDK;
  float*    stg     = (float*)(scratch + 67584) + wave * (16 * 17);
  float*    mergeV  = (float*)(scratch + 76288) + wave * 128;
  unsigned* mergeI  = (unsigned*)(scratch + 80384) + wave * 128;
  float*    wgt     = (float*)(scratch + 84480) + wave * 128;
  unsigned* widx    = (unsigned*)(scratch + 88576) + wave * 128;

  float tv[8]; unsigned tix[8];
#pragma unroll
  for (int k = 0; k < 8; ++k) { tv[k] = -3.0e38f; tix[k] = 0u; }

  __syncthreads();   // scratch region re-purposed

  // stage chunk 0
  {
    const bf16_t* src = keys_bf;
    for (int i = tid; i < 64 * 32; i += 256) {
      int r = i >> 5, c = i & 31;
      cp16_g2lds(kbuf0 + r * LDK + c * 8, src + r * 256 + c * 8);
    }
    async_wait0();
    __syncthreads();
  }

  for (int ch = 0; ch < 64; ++ch) {
    bf16_t* cur = (ch & 1) ? kbuf1 : kbuf0;
    bf16_t* nxt = (ch & 1) ? kbuf0 : kbuf1;
    // kick off async staging of the next chunk; overlaps with WMMA below
    if (ch + 1 < 64) {
      const bf16_t* src = keys_bf + (size_t)(ch + 1) * 64 * 256;
      for (int i = tid; i < 64 * 32; i += 256) {
        int r = i >> 5, c = i & 31;
        cp16_g2lds(nxt + r * LDK + c * 8, src + r * 256 + c * 8);
      }
    }
    const int n0c = ch * 64;
#pragma unroll
    for (int t = 0; t < 4; ++t) {
      v8f acc = {0.f,0.f,0.f,0.f,0.f,0.f,0.f,0.f};
#pragma unroll
      for (int kc = 0; kc < 8; ++kc) {
        v16bf b = load_bf16x16(cur + (t * 16 + lcol) * LDK + kc * 32 + (lhi ? 8 : 0));
        acc = wmma_bf16(a2[kc], b, acc);
      }
      // spill 16x16 sim tile to LDS (C layout: VGPR j -> row j+8*lhi, col lcol)
#pragma unroll
      for (int j = 0; j < 8; ++j) stg[(j + 8 * lhi) * 17 + lcol] = acc[j];
      asm volatile("s_wait_dscnt 0x0" ::: "memory");
      // two lanes per row scan 8 columns each, maintain sorted top-8
      const float* rowp = stg + lcol * 17 + 8 * lhi;
      const unsigned gbase = (unsigned)(n0c + t * 16 + 8 * lhi);
#pragma unroll
      for (int j = 0; j < 8; ++j) {
        float v = rowp[j];
        if (v > tv[7]) {
          tv[7] = v; tix[7] = gbase + j;
#pragma unroll
          for (int s = 7; s > 0; --s) {
            if (tv[s] > tv[s - 1]) {
              float  tf = tv[s]; tv[s] = tv[s - 1]; tv[s - 1] = tf;
              unsigned tu = tix[s]; tix[s] = tix[s - 1]; tix[s - 1] = tu;
            }
          }
        }
      }
    }
    async_wait0();
    __syncthreads();
  }

  // merge lane halves, softmax over final top-8
  if (lhi) {
#pragma unroll
    for (int k = 0; k < 8; ++k) { mergeV[lcol * 8 + k] = tv[k]; mergeI[lcol * 8 + k] = tix[k]; }
  }
  asm volatile("s_wait_dscnt 0x0" ::: "memory");
  if (!lhi) {
#pragma unroll
    for (int k = 0; k < 8; ++k) {
      float v = mergeV[lcol * 8 + k];
      unsigned gi = mergeI[lcol * 8 + k];
      if (v > tv[7]) {
        tv[7] = v; tix[7] = gi;
#pragma unroll
        for (int s = 7; s > 0; --s) {
          if (tv[s] > tv[s - 1]) {
            float  tf = tv[s]; tv[s] = tv[s - 1]; tv[s - 1] = tf;
            unsigned tu = tix[s]; tix[s] = tix[s - 1]; tix[s - 1] = tu;
          }
        }
      }
    }
    float mx = tv[0], ssum = 0.f, w[8];
#pragma unroll
    for (int k = 0; k < 8; ++k) { w[k] = __expf(tv[k] - mx); ssum += w[k]; }
    float inv = 1.0f / ssum;
#pragma unroll
    for (int k = 0; k < 8; ++k) { wgt[lcol * 8 + k] = w[k] * inv; widx[lcol * 8 + k] = tix[k]; }
  }
  asm volatile("s_wait_dscnt 0x0" ::: "memory");

  // gather: retrieved[row] = sum_k w_k * values[idx_k]   (whole wave per row)
  for (int r = 0; r < 16; ++r) {
    float acc8[8];
#pragma unroll
    for (int i = 0; i < 8; ++i) acc8[i] = 0.f;
#pragma unroll
    for (int k = 0; k < 8; ++k) {
      float    wt  = wgt[r * 8 + k];
      unsigned idx = widx[r * 8 + k];
      const float4* vp = (const float4*)(values + (size_t)idx * 256 + lane * 8);
      float4 v0 = vp[0], v1 = vp[1];
      acc8[0] += wt * v0.x; acc8[1] += wt * v0.y; acc8[2] += wt * v0.z; acc8[3] += wt * v0.w;
      acc8[4] += wt * v1.x; acc8[5] += wt * v1.y; acc8[6] += wt * v1.z; acc8[7] += wt * v1.w;
    }
    v8bf o;
#pragma unroll
    for (int i = 0; i < 8; ++i) o[i] = (bf16_t)acc8[i];
    *(v8bf*)(retrieved_bf + (size_t)(rowBlock + mloc + r) * 256 + lane * 8) = o;
  }
}

// ------------------------- fused FFN kernel ---------------------------------
// out = relu([query, retrieved] @ W1 + b1) @ W2 + b2
// grid: 1024 blocks x 128 threads (4 waves); 64 rows per block.

__global__ __launch_bounds__(128)
void k_ffn(const float* __restrict__ query,
           const bf16_t* __restrict__ retrieved_bf,
           const bf16_t* __restrict__ W1T,   // [256][512] = W1^T
           const float* __restrict__ b1,
           const bf16_t* __restrict__ W2T,   // [256][256] = W2^T
           const float* __restrict__ b2,
           float* __restrict__ out)
{
  constexpr int LDH = 520;   // padded 512
  constexpr int LDK = 264;   // padded 256
  __shared__ bf16_t h1 [64 * LDK];   // 33792 B
  __shared__ bf16_t wch[64 * LDH];   // 66560 B (W1T / W2T chunks)

  const int tid  = threadIdx.x;
  const int wave = tid >> 5;
  const int lane = tid & 31;
  const int lhi  = lane >> 4;
  const int lcol = lane & 15;
  const int rowBlock = blockIdx.x * 64;
  const int mloc = wave * 16;

  // A regs (K=512): first 256 from query (f32->bf16), next 256 from retrieved
  v16bf a[16];
  {
    const float*  qp = query        + (size_t)(rowBlock + mloc + lcol) * 256;
    const bf16_t* rp = retrieved_bf + (size_t)(rowBlock + mloc + lcol) * 256;
#pragma unroll
    for (int kc = 0; kc < 8; ++kc)
      a[kc] = load_a_f32(qp + kc * 32 + (lhi ? 8 : 0));
#pragma unroll
    for (int kc = 0; kc < 8; ++kc)
      a[8 + kc] = load_a_lds(rp + kc * 32 + (lhi ? 8 : 0));
  }

  // GEMM 1: h1 = relu(hin @ W1 + b1)
  for (int nch = 0; nch < 4; ++nch) {
    for (int i = tid; i < 64 * 64; i += 128) {          // 64 rows x 64 x 16B
      int r = i >> 6, c = i & 63;
      cp16_g2lds(wch + r * LDH + c * 8,
                 W1T + (size_t)(nch * 64 + r) * 512 + c * 8);
    }
    async_wait0();
    __syncthreads();
#pragma unroll
    for (int t = 0; t < 4; ++t) {
      v8f acc = {0.f,0.f,0.f,0.f,0.f,0.f,0.f,0.f};
#pragma unroll
      for (int kc = 0; kc < 16; ++kc) {
        v16bf b = load_bf16x16(wch + (t * 16 + lcol) * LDH + kc * 32 + (lhi ? 8 : 0));
        acc = wmma_bf16(a[kc], b, acc);
      }
      int ncol = nch * 64 + t * 16 + lcol;
      float bias = b1[ncol];
#pragma unroll
      for (int j = 0; j < 8; ++j) {
        float hv = acc[j] + bias;
        hv = hv > 0.f ? hv : 0.f;
        h1[(mloc + j + 8 * lhi) * LDK + ncol] = (bf16_t)hv;
      }
    }
    __syncthreads();
  }

  // GEMM 2: out = h1 @ W2 + b2
  v16bf a2[8];
  {
    const bf16_t* rp = h1 + (mloc + lcol) * LDK;
#pragma unroll
    for (int kc = 0; kc < 8; ++kc)
      a2[kc] = load_a_lds(rp + kc * 32 + (lhi ? 8 : 0));
  }

  for (int nch = 0; nch < 4; ++nch) {
    for (int i = tid; i < 64 * 32; i += 128) {          // 64 rows x 32 x 16B
      int r = i >> 5, c = i & 31;
      cp16_g2lds(wch + r * LDK + c * 8,
                 W2T + (size_t)(nch * 64 + r) * 256 + c * 8);
    }
    async_wait0();
    __syncthreads();
#pragma unroll
    for (int t = 0; t < 4; ++t) {
      v8f acc = {0.f,0.f,0.f,0.f,0.f,0.f,0.f,0.f};
#pragma unroll
      for (int kc = 0; kc < 8; ++kc) {
        v16bf b = load_bf16x16(wch + (t * 16 + lcol) * LDK + kc * 32 + (lhi ? 8 : 0));
        acc = wmma_bf16(a2[kc], b, acc);
      }
      int ncol = nch * 64 + t * 16 + lcol;
      float bias = b2[ncol];
#pragma unroll
      for (int j = 0; j < 8; ++j)
        out[(size_t)(rowBlock + mloc + j + 8 * lhi) * 256 + ncol] = acc[j] + bias;
    }
    __syncthreads();
  }
}

// ------------------------- launch -------------------------------------------

extern "C" void kernel_launch(void* const* d_in, const int* in_sizes, int n_in,
                              void* d_out, int out_size, void* d_ws, size_t ws_size,
                              hipStream_t stream) {
  const float* query = (const float*)d_in[0];   // (16,4096,256)
  const float* mkeys = (const float*)d_in[1];   // (4096,256)
  const float* mvals = (const float*)d_in[2];   // (4096,256)
  const float* Wq    = (const float*)d_in[3];   // (256,256)
  const float* bq    = (const float*)d_in[4];
  const float* W1    = (const float*)d_in[5];   // (512,256)
  const float* b1    = (const float*)d_in[6];
  const float* W2    = (const float*)d_in[7];   // (256,256)
  const float* b2    = (const float*)d_in[8];
  float* out = (float*)d_out;

  char* ws = (char*)d_ws;
  bf16_t* keys_bf = (bf16_t*)(ws);                                 // 2 MiB
  bf16_t* WqT     = (bf16_t*)(ws + (2u << 20));                    // 128 KiB
  bf16_t* W1T     = (bf16_t*)(ws + (2u << 20) + (128u << 10));     // 256 KiB
  bf16_t* W2T     = (bf16_t*)(ws + (2u << 20) + (384u << 10));     // 128 KiB
  bf16_t* retr    = (bf16_t*)(ws + (2u << 20) + (512u << 10));     // 32 MiB

  k_cvt_bf16<<<(4096 * 256 + 255) / 256, 256, 0, stream>>>(mkeys, keys_bf, 4096 * 256);
  k_transpose_bf16<<<(256 * 256 + 255) / 256, 256, 0, stream>>>(Wq, WqT, 256, 256);
  k_transpose_bf16<<<(512 * 256 + 255) / 256, 256, 0, stream>>>(W1, W1T, 512, 256);
  k_transpose_bf16<<<(256 * 256 + 255) / 256, 256, 0, stream>>>(W2, W2T, 256, 256);

  k_retrieve<<<512, 256, 0, stream>>>(query, keys_bf, mvals, WqT, bq, retr);
  k_ffn<<<1024, 128, 0, stream>>>(query, retr, W1T, b1, W2T, b2, out);
}
